// TransformerStateEncoder_48232482734441
// MI455X (gfx1250) — matrix operand order repaired and checked by reading, not compiled, not run on hardware
//
#include <hip/hip_runtime.h>
#include <cstdint>
#include <cstddef>

// ---------------------------------------------------------------------------
// Types / feature detection
// ---------------------------------------------------------------------------
typedef __attribute__((ext_vector_type(16))) __bf16 v16bf;
typedef __attribute__((ext_vector_type(8)))  __bf16 v8bf;
typedef __attribute__((ext_vector_type(8)))  float  v8f;

// ROCm 7.2 signature: __builtin_amdgcn_global_load_async_to_lds_b128(
//     __attribute__((address_space(1))) int4* src,
//     __attribute__((address_space(3))) int4* dst, imm int offset, imm int cpol)
typedef int i32x4_t __attribute__((vector_size(16)));
typedef __attribute__((address_space(1))) i32x4_t* g128_ptr;
typedef __attribute__((address_space(3))) i32x4_t* l128_ptr;

#ifndef __has_builtin
#define __has_builtin(x) 0
#endif
#if __has_builtin(__builtin_amdgcn_global_load_async_to_lds_b128)
#define HAVE_ASYNC_LDS 1
#else
#define HAVE_ASYNC_LDS 0
#endif
#if __has_builtin(__builtin_amdgcn_s_wait_asynccnt)
#define HAVE_WAIT_ASYNC 1
#else
#define HAVE_WAIT_ASYNC 0
#endif

#define DEV static __device__ __forceinline__

DEV __bf16 f2bf(float f) {
  unsigned u = __builtin_bit_cast(unsigned, f);
  unsigned r = (u + 0x7FFFu + ((u >> 16) & 1u)) >> 16;
  unsigned short s = (unsigned short)r;
  return __builtin_bit_cast(__bf16, s);
}
DEV float bf2f(__bf16 b) {
  unsigned short s = __builtin_bit_cast(unsigned short, b);
  unsigned u = ((unsigned)s) << 16;
  return __builtin_bit_cast(float, u);
}

DEV void wait_async_zero() {
#if HAVE_ASYNC_LDS
#if HAVE_WAIT_ASYNC
  __builtin_amdgcn_s_wait_asynccnt(0);
#else
  asm volatile("s_wait_asynccnt 0x0" ::: "memory");
#endif
#endif
}

// ---------------------------------------------------------------------------
// Problem dims (fixed by the reference)
// ---------------------------------------------------------------------------
constexpr int L_ = 6, T_ = 128, N_ = 32, D_ = 1024, H_ = 16, DH_ = 64, M_ = 128, DI_ = 4096;
constexpr int KL_ = M_ + T_;   // 256
constexpr int TN_ = T_ * N_;   // 4096 rows of h
constexpr int KN_ = KL_ * N_;  // 8192 rows of cat

// ---------------------------------------------------------------------------
// Small elementwise kernels
// ---------------------------------------------------------------------------
__global__ void ep_kernel(const float* __restrict__ masks, int* __restrict__ ep) {
  int nn = threadIdx.x;
  if (nn < N_) {
    int c = 0;
    for (int i = 0; i < T_; ++i) {
      c += (masks[i * N_ + nn] == 0.0f) ? 1 : 0;
      ep[i * N_ + nn] = c;
    }
  }
}

__global__ void add_in_kernel(const float* __restrict__ x, const float* __restrict__ c,
                              float* __restrict__ hf, __bf16* __restrict__ hb, int n) {
  for (int i = blockIdx.x * blockDim.x + threadIdx.x; i < n; i += gridDim.x * blockDim.x) {
    float v = x[i] + c[i];
    hf[i] = v;
    hb[i] = f2bf(v);
  }
}

__global__ void cvt_kernel(const float* __restrict__ in, __bf16* __restrict__ out, int n) {
  for (int i = blockIdx.x * blockDim.x + threadIdx.x; i < n; i += gridDim.x * blockDim.x)
    out[i] = f2bf(in[i]);
}

// Sinusoidal relative-position table R (KL x D), distances KL-1..0
__global__ void rembed_kernel(__bf16* __restrict__ R) {
  int idx = blockIdx.x * 256 + threadIdx.x;
  if (idx >= KL_ * D_) return;
  int jr = idx >> 10;          // D_ == 1024
  int d  = idx & (D_ - 1);
  float pos = (float)(KL_ - 1 - jr);
  int f = (d < D_ / 2) ? d : d - D_ / 2;
  // inv_freq = 10000^(-2f/D) = exp(-(2f/D)*ln 10000)
  float inv = __expf(-(2.0f * (float)f / (float)D_) * 9.210340371976184f);
  float ang = pos * inv;
  R[idx] = f2bf((d < D_ / 2) ? sinf(ang) : cosf(ang));
}

// cat = [mems[l].transpose(1,0,2) * mask0 ; h], flattened (KL*N, D), bf16
__global__ void build_cat_kernel(const float* __restrict__ mems_l, const float* __restrict__ hf,
                                 const float* __restrict__ masks, __bf16* __restrict__ cat) {
  int total = KN_ * D_;
  for (int idx = blockIdx.x * blockDim.x + threadIdx.x; idx < total; idx += gridDim.x * blockDim.x) {
    int d   = idx & (D_ - 1);
    int row = idx >> 10;
    int nn  = row & (N_ - 1);
    int jk  = row >> 5;        // N_ == 32
    float v;
    if (jk < M_) v = mems_l[((size_t)nn * M_ + jk) * D_ + d] * masks[nn];  // masks[0, nn]
    else         v = hf[((size_t)(jk - M_) * N_ + nn) * D_ + d];
    cat[idx] = f2bf(v);
  }
}

// out = LayerNorm(x + delta) -> fp32 master + bf16 copy. One block per row (D=1024).
__global__ __launch_bounds__(256) void add_ln_kernel(
    const float* __restrict__ x, const float* __restrict__ dlt,
    const float* __restrict__ g, const float* __restrict__ b,
    float* __restrict__ outf, __bf16* __restrict__ outb) {
  __shared__ float red[256];
  int row = blockIdx.x, tid = threadIdx.x;
  float vals[4];
  float s = 0.f;
#pragma unroll
  for (int t = 0; t < 4; ++t) {
    int c = tid + t * 256;
    vals[t] = x[(size_t)row * D_ + c] + dlt[(size_t)row * D_ + c];
    s += vals[t];
  }
  red[tid] = s; __syncthreads();
  for (int o = 128; o > 0; o >>= 1) { if (tid < o) red[tid] += red[tid + o]; __syncthreads(); }
  float mu = red[0] * (1.f / D_);
  __syncthreads();
  float s2 = 0.f;
#pragma unroll
  for (int t = 0; t < 4; ++t) { float d = vals[t] - mu; s2 += d * d; }
  red[tid] = s2; __syncthreads();
  for (int o = 128; o > 0; o >>= 1) { if (tid < o) red[tid] += red[tid + o]; __syncthreads(); }
  float var = red[0] * (1.f / D_);
  float rinv = rsqrtf(var + 1e-5f);
#pragma unroll
  for (int t = 0; t < 4; ++t) {
    int c = tid + t * 256;
    float y = (vals[t] - mu) * rinv * g[c] + b[c];
    outf[(size_t)row * D_ + c] = y;
    outb[(size_t)row * D_ + c] = f2bf(y);
  }
}

// ---------------------------------------------------------------------------
// Tiled bf16 WMMA GEMM: C[MxN](f32) = A[MxK](bf16,row) @ B[KxN](bf16,row)
// Block tile 128x128, BK=32, 8 waves of 64x32, v_wmma_f32_16x16x32_bf16.
// Double-buffered LDS; A staged via CDNA5 async global->LDS copies (ASYNCcnt),
// B staged through registers with transpose-on-store (both overlap the WMMAs).
// ---------------------------------------------------------------------------
constexpr int GBM = 128, GBN = 128, GBK = 32;
constexpr int LDA_ = 40;  // padded bf16 row stride (80B: 16B-aligned, conflict-free frags)

DEV v16bf combine16(v8bf lo, v8bf hi) {
  return __builtin_shufflevector(lo, hi, 0, 1, 2, 3, 4, 5, 6, 7, 8, 9, 10, 11, 12, 13, 14, 15);
}

// Stage a 128x32 bf16 A tile: 512 chunks of 8 elems (16B), 2 per thread.
DEV void stageA(const __bf16* __restrict__ A, __bf16* dst, int tid, int mBase, int K, int k0) {
#pragma unroll
  for (int it = 0; it < 2; ++it) {
    int c = tid + it * 256;
    int row = c >> 2, kc = (c & 3) * 8;
    const __bf16* g = A + (size_t)(mBase + row) * K + k0 + kc;
    __bf16* l = dst + row * LDA_ + kc;
#if HAVE_ASYNC_LDS
    __builtin_amdgcn_global_load_async_to_lds_b128((g128_ptr)g, (l128_ptr)l, 0, 0);
#else
    *(v8bf*)l = *(const v8bf*)g;
#endif
  }
}

DEV void loadBregs(const __bf16* __restrict__ B, v8bf* regs, int tid, int nBase, int N, int k0) {
#pragma unroll
  for (int it = 0; it < 2; ++it) {
    int c = tid + it * 256;
    int kk = c >> 4, nc = (c & 15) * 8;
    regs[it] = *(const v8bf*)(B + (size_t)(k0 + kk) * N + nBase + nc);
  }
}

DEV void storeBlds(__bf16* sB, const v8bf* regs, int tid) {
#pragma unroll
  for (int it = 0; it < 2; ++it) {
    int c = tid + it * 256;
    int kk = c >> 4, nc = (c & 15) * 8;
#pragma unroll
    for (int j = 0; j < 8; ++j) sB[(nc + j) * LDA_ + kk] = regs[it][j];
  }
}

__global__ __launch_bounds__(256) void gemm_bf16_kernel(
    const __bf16* __restrict__ A, const __bf16* __restrict__ B,
    const float* __restrict__ bias, float* __restrict__ Cf, __bf16* __restrict__ Cb,
    int M, int N, int K, int relu) {
  __shared__ __bf16 sA[2][GBM * LDA_];  // [row][k], double buffered
  __shared__ __bf16 sB[2][GBN * LDA_];  // transposed [n][k], double buffered

  const int tid = threadIdx.x;
  const int wave = tid >> 5, lane = tid & 31;
  const int wm = wave >> 2, wn = wave & 3;  // 2x4 wave grid -> 64x32 per wave
  const int mBase = blockIdx.y * GBM;
  const int nBase = blockIdx.x * GBN;

  v8f acc[4][2];
#pragma unroll
  for (int mi = 0; mi < 4; ++mi)
#pragma unroll
    for (int ni = 0; ni < 2; ++ni)
#pragma unroll
      for (int r = 0; r < 8; ++r) acc[mi][ni][r] = 0.f;

  const int halfsel = lane >> 4;       // 0: lanes 0-15, 1: lanes 16-31
  const int rlane = lane & 15;
  const int ka0 = halfsel ? 8 : 0;     // A frag v0-3 K base
  const int ka1 = ka0 + 16;            // A frag v4-7 K base
  const int kb0 = halfsel ? 8 : 0;     // B frag v0-3 K base (v4-7 at +16), same as A

  // prologue: tile 0 in flight
  v8bf bregs[2];
  stageA(A, sA[0], tid, mBase, K, 0);
  loadBregs(B, bregs, tid, nBase, N, 0);

  const int nt = K / GBK;
  for (int t = 0; t < nt; ++t) {
    const int cur = t & 1;
    __bf16* sAc = sA[cur];
    __bf16* sBc = sB[cur];

    wait_async_zero();              // this wave's async writes into sA[cur] done
    storeBlds(sBc, bregs, tid);     // B tile t into LDS (transposed)
    __syncthreads();                // tile t fully staged & visible to all waves

    if (t + 1 < nt) {               // kick off tile t+1 while computing tile t
      stageA(A, sA[cur ^ 1], tid, mBase, K, (t + 1) * GBK);
      loadBregs(B, bregs, tid, nBase, N, (t + 1) * GBK);
    }

    v16bf afrag[4];
#pragma unroll
    for (int mi = 0; mi < 4; ++mi) {
      int row = wm * 64 + mi * 16 + rlane;
      v8bf lo = *(const v8bf*)(&sAc[row * LDA_ + ka0]);
      v8bf hi = *(const v8bf*)(&sAc[row * LDA_ + ka1]);
      afrag[mi] = combine16(lo, hi);
    }
    v16bf bfrag[2];
#pragma unroll
    for (int ni = 0; ni < 2; ++ni) {
      int col = wn * 32 + ni * 16 + rlane;
      v8bf lo = *(const v8bf*)(&sBc[col * LDA_ + kb0]);
      v8bf hi = *(const v8bf*)(&sBc[col * LDA_ + kb0 + 16]);
      bfrag[ni] = combine16(lo, hi);
    }
#pragma unroll
    for (int mi = 0; mi < 4; ++mi)
#pragma unroll
      for (int ni = 0; ni < 2; ++ni)
        acc[mi][ni] = __builtin_amdgcn_wmma_f32_16x16x32_bf16(
            false, afrag[mi], false, bfrag[ni], (short)0, acc[mi][ni], false, false);
    __syncthreads();                // all waves done reading tile t buffers
  }

  // ---- epilogue: C layout -> lane 0-15: M=r, N=lane; lane 16-31: M=8+r ----
  const int rowOff = halfsel * 8;
#pragma unroll
  for (int mi = 0; mi < 4; ++mi)
#pragma unroll
    for (int ni = 0; ni < 2; ++ni) {
      int col = nBase + wn * 32 + ni * 16 + rlane;
      float bv = bias ? bias[col] : 0.f;
#pragma unroll
      for (int r = 0; r < 8; ++r) {
        int row = mBase + wm * 64 + mi * 16 + rowOff + r;
        float v = acc[mi][ni][r] + bv;
        if (relu) v = fmaxf(v, 0.f);
        if (Cf) Cf[(size_t)row * N + col] = v;
        if (Cb) Cb[(size_t)row * N + col] = f2bf(v);
      }
    }
}

// ---------------------------------------------------------------------------
// Attention: one block per (env, head). Rel-shifted TXL scores + episode mask
// + softmax + attn@v, all from LDS-staged bf16 tiles.
// ---------------------------------------------------------------------------
constexpr size_t ATTN_SMEM =
    (size_t)(2 * T_ * DH_ + 3 * KL_ * DH_) * 2 + (size_t)8 * KL_ * 4 + (size_t)T_ * 4;

__global__ __launch_bounds__(256) void attn_kernel(
    const float* __restrict__ q, const float* __restrict__ k, const float* __restrict__ v,
    const float* __restrict__ r, const float* __restrict__ rwb, const float* __restrict__ rrb,
    const int* __restrict__ ep, __bf16* __restrict__ obf) {
  extern __shared__ char smem[];
  __bf16* sQw = (__bf16*)smem;                 // T x DH : q + r_w_bias
  __bf16* sQr = sQw + T_ * DH_;                // T x DH : q + r_r_bias
  __bf16* sK  = sQr + T_ * DH_;                // KL x DH
  __bf16* sR  = sK + KL_ * DH_;                // KL x DH
  __bf16* sV  = sR + KL_ * DH_;                // KL x DH
  float*  sAt = (float*)(sV + KL_ * DH_);      // 8 waves x KL
  int*    sEp = (int*)(sAt + 8 * KL_);         // T

  const int nn = blockIdx.x >> 4;   // env
  const int hh = blockIdx.x & 15;   // head
  const int tid = threadIdx.x;

  for (int e = tid; e < T_ * DH_; e += 256) {
    int i = e >> 6, d = e & 63;
    float qv = q[((size_t)(i * N_ + nn)) * (H_ * DH_) + hh * DH_ + d];
    sQw[e] = f2bf(qv + rwb[hh * DH_ + d]);
    sQr[e] = f2bf(qv + rrb[hh * DH_ + d]);
  }
  for (int e = tid; e < KL_ * DH_; e += 256) {
    int j = e >> 6, d = e & 63;
    size_t kvrow = ((size_t)(j * N_ + nn)) * (H_ * DH_) + hh * DH_ + d;
    sK[e] = f2bf(k[kvrow]);
    sV[e] = f2bf(v[kvrow]);
    sR[e] = f2bf(r[(size_t)j * (H_ * DH_) + hh * DH_ + d]);
  }
  for (int e = tid; e < T_; e += 256) sEp[e] = ep[e * N_ + nn];
  __syncthreads();

  const int wave = tid >> 5, lane = tid & 31;
  const float scale = 0.125f;  // 1/sqrt(64)

  for (int i = wave; i < T_; i += 8) {
    int epi = sEp[i];
    float sc[8];
#pragma unroll
    for (int jj = 0; jj < 8; ++jj) {
      int j = lane + jj * 32;
      float ac = 0.f;
      for (int d = 0; d < DH_; ++d) ac += bf2f(sQw[i * DH_ + d]) * bf2f(sK[j * DH_ + d]);
      float bd = 0.f;
      int jr = j + (T_ - 1) - i;  // rel_shift: shifted[i,j] = raw[i, j + t-1 - i]
      if (jr >= 0 && jr < KL_)
        for (int d = 0; d < DH_; ++d) bd += bf2f(sQr[i * DH_ + d]) * bf2f(sR[jr * DH_ + d]);
      bool allow = (j < M_) ? (epi == 0) : ((j - M_) <= i && sEp[j - M_] == epi);
      sc[jj] = allow ? (ac + bd) * scale : -1e9f;
    }
    float mx = sc[0];
#pragma unroll
    for (int jj = 1; jj < 8; ++jj) mx = fmaxf(mx, sc[jj]);
    for (int o = 16; o > 0; o >>= 1) mx = fmaxf(mx, __shfl_xor(mx, o, 32));
    float sum = 0.f;
#pragma unroll
    for (int jj = 0; jj < 8; ++jj) { sc[jj] = __expf(sc[jj] - mx); sum += sc[jj]; }
    for (int o = 16; o > 0; o >>= 1) sum += __shfl_xor(sum, o, 32);
    float inv = 1.f / sum;
#pragma unroll
    for (int jj = 0; jj < 8; ++jj) sAt[wave * KL_ + lane + jj * 32] = sc[jj] * inv;
    __builtin_amdgcn_wave_barrier();
#pragma unroll
    for (int dd = 0; dd < 2; ++dd) {
      int d = lane * 2 + dd;
      float o = 0.f;
      for (int j = 0; j < KL_; ++j) o += sAt[wave * KL_ + j] * bf2f(sV[j * DH_ + d]);
      obf[((size_t)(i * N_ + nn)) * (H_ * DH_) + hh * DH_ + d] = f2bf(o);
    }
    __builtin_amdgcn_wave_barrier();
  }
}

// ---------------------------------------------------------------------------
// Host-side orchestration
// ---------------------------------------------------------------------------
static void launch_gemm(const __bf16* A, const __bf16* B, const float* bias, float* Cf,
                        __bf16* Cb, int M, int N, int K, int relu, hipStream_t s) {
  dim3 grid(N / GBN, M / GBM);
  gemm_bf16_kernel<<<grid, 256, 0, s>>>(A, B, bias, Cf, Cb, M, N, K, relu);
}

extern "C" void kernel_launch(void* const* d_in, const int* in_sizes, int n_in,
                              void* d_out, int out_size, void* d_ws, size_t ws_size,
                              hipStream_t stream) {
  (void)in_sizes; (void)n_in; (void)out_size; (void)ws_size;

  const float* x     = (const float*)d_in[0];
  const float* ctx   = (const float*)d_in[1];
  const float* mems  = (const float*)d_in[2];
  const float* masks = (const float*)d_in[3];
  const float* Wq    = (const float*)d_in[4];
  const float* Wk    = (const float*)d_in[5];
  const float* Wv    = (const float*)d_in[6];
  const float* Wr    = (const float*)d_in[7];
  const float* Wo    = (const float*)d_in[8];
  const float* W1    = (const float*)d_in[9];
  const float* b1    = (const float*)d_in[10];
  const float* W2    = (const float*)d_in[11];
  const float* b2    = (const float*)d_in[12];
  const float* ln1g  = (const float*)d_in[13];
  const float* ln1b  = (const float*)d_in[14];
  const float* ln2g  = (const float*)d_in[15];
  const float* ln2b  = (const float*)d_in[16];
  const float* rwb   = (const float*)d_in[17];
  const float* rrb   = (const float*)d_in[18];

  // --- workspace carve (256B aligned) ---
  char* p = (char*)d_ws;
  auto carve = [&](size_t bytes) -> void* {
    void* r = (void*)p;
    p += (bytes + 255) & ~(size_t)255;
    return r;
  };
  int*    ep   = (int*)   carve((size_t)TN_ * 4);
  float*  hF   = (float*) carve((size_t)TN_ * D_ * 4);
  __bf16* hB   = (__bf16*)carve((size_t)TN_ * D_ * 2);
  __bf16* Rbf  = (__bf16*)carve((size_t)KL_ * D_ * 2);
  __bf16* catB = (__bf16*)carve((size_t)KN_ * D_ * 2);
  float*  qF   = (float*) carve((size_t)TN_ * D_ * 4);       // also GEMM scratch
  float*  kF   = (float*) carve((size_t)KN_ * D_ * 4);       // reused as FFN hidden (bf16)
  float*  vF   = (float*) carve((size_t)KN_ * D_ * 4);
  float*  rF   = (float*) carve((size_t)KL_ * D_ * 4);
  __bf16* oB   = (__bf16*)carve((size_t)TN_ * D_ * 2);
  __bf16* WqB  = (__bf16*)carve((size_t)D_ * H_ * DH_ * 2);
  __bf16* WkB  = (__bf16*)carve((size_t)D_ * H_ * DH_ * 2);
  __bf16* WvB  = (__bf16*)carve((size_t)D_ * H_ * DH_ * 2);
  __bf16* WrB  = (__bf16*)carve((size_t)D_ * H_ * DH_ * 2);
  __bf16* WoB  = (__bf16*)carve((size_t)H_ * DH_ * D_ * 2);
  __bf16* W1B  = (__bf16*)carve((size_t)D_ * DI_ * 2);
  __bf16* W2B  = (__bf16*)carve((size_t)DI_ * D_ * 2);
  __bf16* fB   = (__bf16*)kF;  // FFN hidden (4096x4096 bf16) reuses kF (32MB)

  const int EW = D_ * H_ * DH_;  // 1M elems per square weight

  ep_kernel<<<1, 32, 0, stream>>>(masks, ep);
  add_in_kernel<<<2048, 256, 0, stream>>>(x, ctx, hF, hB, TN_ * D_);
  rembed_kernel<<<(KL_ * D_) / 256, 256, 0, stream>>>(Rbf);

  for (int l = 0; l < L_; ++l) {
    // per-layer weight conversion fp32 -> bf16
    cvt_kernel<<<2048, 256, 0, stream>>>(Wq + (size_t)l * EW, WqB, EW);
    cvt_kernel<<<2048, 256, 0, stream>>>(Wk + (size_t)l * EW, WkB, EW);
    cvt_kernel<<<2048, 256, 0, stream>>>(Wv + (size_t)l * EW, WvB, EW);
    cvt_kernel<<<2048, 256, 0, stream>>>(Wr + (size_t)l * EW, WrB, EW);
    cvt_kernel<<<2048, 256, 0, stream>>>(Wo + (size_t)l * EW, WoB, EW);
    cvt_kernel<<<4096, 256, 0, stream>>>(W1 + (size_t)l * D_ * DI_, W1B, D_ * DI_);
    cvt_kernel<<<4096, 256, 0, stream>>>(W2 + (size_t)l * DI_ * D_, W2B, DI_ * D_);

    build_cat_kernel<<<8192, 256, 0, stream>>>(mems + (size_t)l * N_ * M_ * D_, hF, masks, catB);

    // projections
    launch_gemm(hB,   WqB, nullptr, qF, nullptr, TN_, H_ * DH_, D_, 0, stream);
    launch_gemm(catB, WkB, nullptr, kF, nullptr, KN_, H_ * DH_, D_, 0, stream);
    launch_gemm(catB, WvB, nullptr, vF, nullptr, KN_, H_ * DH_, D_, 0, stream);
    launch_gemm(Rbf,  WrB, nullptr, rF, nullptr, KL_, H_ * DH_, D_, 0, stream);

    // attention (one block per env x head)
    attn_kernel<<<N_ * H_, 256, ATTN_SMEM, stream>>>(qF, kF, vF, rF, rwb, rrb, ep, oB);

    // output projection + residual LN1
    launch_gemm(oB, WoB, nullptr, qF, nullptr, TN_, D_, H_ * DH_, 0, stream);
    add_ln_kernel<<<TN_, 256, 0, stream>>>(hF, qF, ln1g + (size_t)l * D_, ln1b + (size_t)l * D_, hF, hB);

    // FFN (relu fused, bf16 hidden), residual LN2
    launch_gemm(hB, W1B, b1 + (size_t)l * DI_, nullptr, fB, TN_, DI_, D_, 1, stream);
    launch_gemm(fB, W2B, b2 + (size_t)l * D_,  qF, nullptr, TN_, D_, DI_, 0, stream);
    add_ln_kernel<<<TN_, 256, 0, stream>>>(hF, qF, ln2g + (size_t)l * D_, ln2b + (size_t)l * D_, hF, hB);
  }

  (void)hipMemcpyAsync(d_out, hF, (size_t)TN_ * D_ * sizeof(float),
                       hipMemcpyDeviceToDevice, stream);
}
